// MultiheadAttention_73882027426164
// MI455X (gfx1250) — compile-verified, matrix operand order checked by reading, not run on hardware
//
#include <hip/hip_runtime.h>
#include <hip/hip_bf16.h>

// ---------- types ----------
typedef __attribute__((ext_vector_type(16))) __bf16 v16bf;
typedef __attribute__((ext_vector_type(8)))  __bf16 v8bf;
typedef __attribute__((ext_vector_type(8)))  float  v8f;

#define N_HEADS 12
#define HID     768
#define BATCH   8
#define SEQ     512
#define NH      (N_HEADS * HID)        // 9216
#define D3      (3 * NH)               // 27648
#define MROWS   (BATCH * SEQ)          // 4096

// ---------- bf16 helpers (bit-exact round-to-nearest-even split) ----------
__device__ __forceinline__ unsigned short f2bf(float f) {
    unsigned u = __float_as_uint(f);
    u += 0x7FFFu + ((u >> 16) & 1u);
    return (unsigned short)(u >> 16);
}
__device__ __forceinline__ float bf2f(unsigned short h) {
    return __uint_as_float(((unsigned)h) << 16);
}

// Load one 16x32 bf16 A/B fragment for this lane from an LDS tile stored as
// [row][32] ushort (row = M for A, row = N for B).  Per CDNA5 ISA layout:
// lanes 0-15 hold K = {0..7, 16..23}, lanes 16-31 hold K = {8..15, 24..31}.
__device__ __forceinline__ v16bf ldfrag(const unsigned short* p) {
    v8bf lo = *(const v8bf*)(p);        // K = kb .. kb+7
    v8bf hi = *(const v8bf*)(p + 16);   // K = kb+16 .. kb+23
    return __builtin_shufflevector(lo, hi,
        0,1,2,3,4,5,6,7,8,9,10,11,12,13,14,15);
}

__device__ __forceinline__ v8f wmma_bf16(v16bf a, v16bf b, v8f c) {
    return __builtin_amdgcn_wmma_f32_16x16x32_bf16(
        /*neg_a=*/false, a, /*neg_b=*/false, b,
        /*c_mod=*/(short)0, c, /*reuse_a=*/false, /*reuse_b=*/false);
}

// ---------------------------------------------------------------------------
// Tiled WMMA GEMM core.
//   TRANSB == false : C[M,N] = A[M,K] * B[N,K]^T (+bias)   (NT)
//   TRANSB == true  : C[M,N] = A[M,K] * B[K,N]   (+bias)   (NN)
// fp32 inputs are split into bf16 hi+lo; 3 WMMAs per tile recover ~fp32.
// Block: 256 threads, 128x128 macrotile, grid (N/128, M/128, Z).
// ---------------------------------------------------------------------------
template<bool TRANSB>
__device__ __forceinline__ void gemm_core(
    const float* __restrict__ A, int lda,
    const float* __restrict__ B, int ldb,
    float* __restrict__ C, int ldc,
    int K, const float* __restrict__ bias)
{
    __shared__ alignas(16) unsigned short lAh[128 * 32];
    __shared__ alignas(16) unsigned short lAl[128 * 32];
    __shared__ alignas(16) unsigned short lBh[128 * 32];
    __shared__ alignas(16) unsigned short lBl[128 * 32];

    const int tid  = threadIdx.x;
    const int lane = tid & 31;
    const int wid  = tid >> 5;      // 0..7
    const int wm   = wid & 3;       // 4 waves along M (32 rows each)
    const int wn   = wid >> 2;      // 2 waves along N (64 cols each)
    const int m0   = blockIdx.y * 128;
    const int n0   = blockIdx.x * 128;

    const int lrow = lane & 15;          // M (A/C) or N (B) index within tile
    const int kb   = (lane >> 4) << 3;   // 0 or 8: lane-group K base

    v8f acc[2][4];
#pragma unroll
    for (int s = 0; s < 2; ++s)
#pragma unroll
        for (int t = 0; t < 4; ++t)
            acc[s][t] = (v8f)(0.0f);

    for (int k0 = 0; k0 < K; k0 += 32) {
        // ---- stage A tile (128 x 32) as hi/lo bf16 ----
#pragma unroll
        for (int i = 0; i < 4; ++i) {
            int idx4 = tid + i * 256;            // 0..1023 float4 slots
            int r    = idx4 >> 3;                // 0..127
            int c4   = (idx4 & 7) << 2;          // 0,4,...,28
            const float4 v = *(const float4*)(A + (size_t)(m0 + r) * lda + k0 + c4);
            float f[4] = {v.x, v.y, v.z, v.w};
#pragma unroll
            for (int j = 0; j < 4; ++j) {
                unsigned short h = f2bf(f[j]);
                lAh[r * 32 + c4 + j] = h;
                lAl[r * 32 + c4 + j] = f2bf(f[j] - bf2f(h));
            }
        }
        // ---- stage B tile as (N-row, K-col) hi/lo bf16 ----
        if (!TRANSB) {
            // B is (N,K) row-major: direct copy
#pragma unroll
            for (int i = 0; i < 4; ++i) {
                int idx4 = tid + i * 256;
                int r    = idx4 >> 3;            // n within macrotile
                int c4   = (idx4 & 7) << 2;
                const float4 v = *(const float4*)(B + (size_t)(n0 + r) * ldb + k0 + c4);
                float f[4] = {v.x, v.y, v.z, v.w};
#pragma unroll
                for (int j = 0; j < 4; ++j) {
                    unsigned short h = f2bf(f[j]);
                    lBh[r * 32 + c4 + j] = h;
                    lBl[r * 32 + c4 + j] = f2bf(f[j] - bf2f(h));
                }
            }
        } else {
            // B is (K,N) row-major: transpose into LDS (coalesced over N)
#pragma unroll
            for (int i = 0; i < 4; ++i) {
                int idx4 = tid + i * 256;
                int kk   = idx4 >> 5;            // 0..31
                int n4   = (idx4 & 31) << 2;     // 0..124
                const float4 v = *(const float4*)(B + (size_t)(k0 + kk) * ldb + n0 + n4);
                float f[4] = {v.x, v.y, v.z, v.w};
#pragma unroll
                for (int j = 0; j < 4; ++j) {
                    unsigned short h = f2bf(f[j]);
                    lBh[(n4 + j) * 32 + kk] = h;
                    lBl[(n4 + j) * 32 + kk] = f2bf(f[j] - bf2f(h));
                }
            }
        }
        __syncthreads();

        // ---- WMMA compute: 2x4 tiles x 3 passes (hi*hi + hi*lo + lo*hi) ----
        v16bf ah[2], al[2];
#pragma unroll
        for (int s = 0; s < 2; ++s) {
            int row = wm * 32 + s * 16 + lrow;
            ah[s] = ldfrag(lAh + row * 32 + kb);
            al[s] = ldfrag(lAl + row * 32 + kb);
        }
#pragma unroll
        for (int t = 0; t < 4; ++t) {
            int col = wn * 64 + t * 16 + lrow;
            v16bf bh = ldfrag(lBh + col * 32 + kb);
            v16bf bl = ldfrag(lBl + col * 32 + kb);
#pragma unroll
            for (int s = 0; s < 2; ++s) {
                acc[s][t] = wmma_bf16(ah[s], bh, acc[s][t]);
                acc[s][t] = wmma_bf16(ah[s], bl, acc[s][t]);
                acc[s][t] = wmma_bf16(al[s], bh, acc[s][t]);
            }
        }
        __syncthreads();
    }

    // ---- epilogue: C/D layout -> lanes 0-15 M=r, lanes 16-31 M=r+8 ----
    const int rbase = (lane >> 4) << 3;
#pragma unroll
    for (int s = 0; s < 2; ++s)
#pragma unroll
        for (int t = 0; t < 4; ++t) {
            int gn = n0 + wn * 64 + t * 16 + lrow;
            float bval = bias ? bias[gn] : 0.0f;
#pragma unroll
            for (int r = 0; r < 8; ++r) {
                int gm = m0 + wm * 32 + s * 16 + rbase + r;
                C[(size_t)gm * ldc + gn] = acc[s][t][r] + bval;
            }
        }
}

// ---------------------------------------------------------------------------
// Stage kernels
// ---------------------------------------------------------------------------
__global__ __launch_bounds__(256) void k_gemm_qkv(
    const float* __restrict__ x, const float* __restrict__ w,
    const float* __restrict__ bias, float* __restrict__ out)
{
    gemm_core<false>(x, HID, w, HID, out, D3, HID, bias);
}

__global__ __launch_bounds__(256) void k_scores(
    const float* __restrict__ qkv, float* __restrict__ scores)
{
    int z = blockIdx.z;                 // z = b*12 + head
    int b = z / N_HEADS, h = z % N_HEADS;
    const float* A = qkv + (size_t)b * SEQ * D3 + (size_t)h * HID;       // Q rows
    const float* B = A + NH;                                             // K rows
    float* C = scores + (size_t)z * SEQ * SEQ;
    gemm_core<false>(A, D3, B, D3, C, SEQ, HID, nullptr);
}

__global__ __launch_bounds__(256) void k_softmax(
    float* __restrict__ scores, const float* __restrict__ mask)
{
    // grid: (SEQ query rows, 96 batch*head)
    int q = blockIdx.x, z = blockIdx.y, b = z / N_HEADS;
    float* row = scores + ((size_t)z * SEQ + q) * SEQ;
    const float* mrow = mask + (size_t)b * SEQ;
    const int t = threadIdx.x;
    const float scale = 0.03608439182435161f;   // 1/sqrt(768)

    float v0 = row[t]       * scale + mrow[t];
    float v1 = row[t + 256] * scale + mrow[t + 256];

    __shared__ float red[256];
    red[t] = fmaxf(v0, v1);
    __syncthreads();
#pragma unroll
    for (int s = 128; s > 0; s >>= 1) {
        if (t < s) red[t] = fmaxf(red[t], red[t + s]);
        __syncthreads();
    }
    float m = red[0];
    __syncthreads();

    float e0 = __expf(v0 - m);
    float e1 = __expf(v1 - m);
    red[t] = e0 + e1;
    __syncthreads();
#pragma unroll
    for (int s = 128; s > 0; s >>= 1) {
        if (t < s) red[t] += red[t + s];
        __syncthreads();
    }
    float inv = 1.0f / red[0];
    row[t]       = e0 * inv;
    row[t + 256] = e1 * inv;
}

__global__ __launch_bounds__(256) void k_av(
    const float* __restrict__ qkv, const float* __restrict__ probs,
    float* __restrict__ av)
{
    int z = blockIdx.z;
    int b = z / N_HEADS, h = z % N_HEADS;
    const float* A = probs + (size_t)z * SEQ * SEQ;                       // (512,512)
    const float* B = qkv + (size_t)b * SEQ * D3 + 2 * NH + (size_t)h * HID; // V rows (K=512, ld=D3)
    // write straight into (b, s, n*h) layout for the output projection
    float* C = av + (size_t)b * SEQ * NH + (size_t)h * HID;
    gemm_core<true>(A, SEQ, B, D3, C, NH, SEQ, nullptr);
}

__global__ __launch_bounds__(256) void k_proj(
    const float* __restrict__ av, const float* __restrict__ w_o,
    const float* __restrict__ b_o, float* __restrict__ out)
{
    gemm_core<false>(av, NH, w_o, NH, out, HID, NH, b_o);
}

// ---------------------------------------------------------------------------
extern "C" void kernel_launch(void* const* d_in, const int* in_sizes, int n_in,
                              void* d_out, int out_size, void* d_ws, size_t ws_size,
                              hipStream_t stream) {
    (void)in_sizes; (void)n_in; (void)out_size; (void)ws_size;
    const float* x     = (const float*)d_in[0];
    const float* mask  = (const float*)d_in[1];
    const float* w_qkv = (const float*)d_in[2];
    const float* b_qkv = (const float*)d_in[3];
    const float* w_o   = (const float*)d_in[4];
    const float* b_o   = (const float*)d_in[5];
    float* out = (float*)d_out;

    // workspace partition (floats): qkv | scores | av
    float* qkv    = (float*)d_ws;
    float* scores = qkv    + (size_t)BATCH * SEQ * D3;        // 113,246,208 f
    float* av     = scores + (size_t)BATCH * N_HEADS * SEQ * SEQ; // +25,165,824 f

    dim3 blk(256);
    k_gemm_qkv<<<dim3(D3 / 128, MROWS / 128, 1), blk, 0, stream>>>(x, w_qkv, b_qkv, qkv);
    k_scores  <<<dim3(SEQ / 128, SEQ / 128, BATCH * N_HEADS), blk, 0, stream>>>(qkv, scores);
    k_softmax <<<dim3(SEQ, BATCH * N_HEADS), blk, 0, stream>>>(scores, mask);
    k_av      <<<dim3(HID / 128, SEQ / 128, BATCH * N_HEADS), blk, 0, stream>>>(qkv, scores, av);
    k_proj    <<<dim3(HID / 128, MROWS / 128, 1), blk, 0, stream>>>(av, w_o, b_o, out);
}
